// Quantize_15745350108065
// MI455X (gfx1250) — matrix operand order repaired and checked by reading, not compiled, not run on hardware
//
#include <hip/hip_runtime.h>

#define K_DIM 256
#define N_EMB 4096
#define N_TOK 65536
#define DIFF_OFF 16777216
#define IND_OFF  16777217

typedef __attribute__((ext_vector_type(16))) __bf16 v16bf;
typedef __attribute__((ext_vector_type(8)))  float  v8f;
typedef __attribute__((ext_vector_type(4)))  float  v4f;

union BFrag { v16bf v; unsigned short s[16]; };

static __device__ __forceinline__ unsigned short f2bf(float f) {
    union { float f; unsigned u; } c; c.f = f;
    unsigned u = c.u;
    // round-to-nearest-even f32 -> bf16
    unsigned r = (u + 0x7FFFu + ((u >> 16) & 1u)) >> 16;
    return (unsigned short)r;
}

// ---------------------------------------------------------------------------
// Prep 1: transpose codebook [K, E] -> code-major [E, K] in f32 (for exact
// gathers) and bf16 (WMMA B operand source).
// ---------------------------------------------------------------------------
__global__ __launch_bounds__(256) void vq_prep_transpose(
    const float* __restrict__ embed, float* __restrict__ embT,
    unsigned short* __restrict__ eb)
{
    int i = blockIdx.x * 256 + threadIdx.x;   // 0 .. K_DIM*N_EMB-1
    int k = i >> 12;                          // / N_EMB
    int e = i & (N_EMB - 1);
    float v = embed[i];
    embT[(size_t)e * K_DIM + k] = v;
    eb[(size_t)e * K_DIM + k]   = f2bf(v);
}

// ---------------------------------------------------------------------------
// Prep 2: per-code squared norms cn[e] = sum_k embed[k][e]^2 (coalesced over e)
// ---------------------------------------------------------------------------
__global__ __launch_bounds__(256) void vq_prep_norm(
    const float* __restrict__ embed, float* __restrict__ cn)
{
    int e = blockIdx.x * 256 + threadIdx.x;   // 16 blocks x 256 = 4096
    float s = 0.0f;
    for (int k = 0; k < K_DIM; ++k) {
        float v = embed[(size_t)k * N_EMB + e];
        s = fmaf(v, v, s);
    }
    cn[e] = s;
}

// ---------------------------------------------------------------------------
// Main: each wave = 32 token rows (two 16-row A-fragment sets resident in
// VGPRs). Sweep all 4096 codes in 16-wide tiles: each B fragment feeds TWO
// v_wmma_f32_16x16x32_bf16 (independent accumulators -> back-to-back XDL
// issue, 16 WMMA per 17 loads per tile). Running argmin of
// (||e||^2 - 2 x.e), then exact-f32 gather of winning rows + MSE partial.
// ---------------------------------------------------------------------------
__global__ __launch_bounds__(256) void vq_main(
    const float* __restrict__ x, const unsigned short* __restrict__ eb,
    const float* __restrict__ cn, const float* __restrict__ embT,
    float* __restrict__ qout, float* __restrict__ indout,
    float* __restrict__ partials)
{
    const int tid  = threadIdx.x;
    const int wave = tid >> 5;
    const int lane = tid & 31;
    const int half = lane >> 4;   // 0: lanes 0-15, 1: lanes 16-31
    const int l    = lane & 15;
    const int r0   = (blockIdx.x * 8 + wave) * 32;   // 32 rows per wave

    // ---- Build 2 x 8 bf16 A fragments (32 rows x K=256) in contiguous k order (B uses the same):
    // lanes 0-15: elems 0-15 = K kb+0..15
    // lanes 16-31: elems 0-15 = K kb+16..31
    BFrag afrag[2][8];
    #pragma unroll
    for (int s = 0; s < 2; ++s) {
        const float* xr = x + (size_t)(r0 + s * 16 + l) * K_DIM;
        #pragma unroll
        for (int f = 0; f < 8; ++f) {
            const int b0 = 32 * f + half * 16;
            v4f a0 = *(const v4f*)(xr + b0);
            v4f a1 = *(const v4f*)(xr + b0 + 4);
            v4f a2 = *(const v4f*)(xr + b0 + 8);
            v4f a3 = *(const v4f*)(xr + b0 + 12);
            #pragma unroll
            for (int i = 0; i < 4; ++i) {
                afrag[s][f].s[i]      = f2bf(a0[i]);
                afrag[s][f].s[4 + i]  = f2bf(a1[i]);
                afrag[s][f].s[8 + i]  = f2bf(a2[i]);
                afrag[s][f].s[12 + i] = f2bf(a3[i]);
            }
        }
    }

    float best[2][8];
    int   bidx[2][8];
    #pragma unroll
    for (int s = 0; s < 2; ++s)
        #pragma unroll
        for (int r = 0; r < 8; ++r) { best[s][r] = 3.402823466e38f; bidx[s][r] = 0; }

    // ---- Sweep all codes in 16-column tiles
    #pragma unroll 1
    for (int t = 0; t < N_EMB / 16; ++t) {
        const int   col = t * 16 + l;          // this lane's code column
        const float cs  = cn[col];
        // B-layout: lanes 0-15 hold K kb+0..15, lanes 16-31 K kb+16..31 (col = l)
        const unsigned short* bp = eb + (size_t)col * K_DIM + half * 16;

        v8f acc0 = {};
        v8f acc1 = {};
        #pragma unroll
        for (int f = 0; f < 8; ++f) {
            v16bf bf = *(const v16bf*)(bp + 32 * f);
            acc0 = __builtin_amdgcn_wmma_f32_16x16x32_bf16(
                false, afrag[0][f].v, false, bf, (short)0, acc0, false, false);
            acc1 = __builtin_amdgcn_wmma_f32_16x16x32_bf16(
                false, afrag[1][f].v, false, bf, (short)0, acc1, false, false);
        }
        #pragma unroll
        for (int r = 0; r < 8; ++r) {
            float sc0 = fmaf(-2.0f, acc0[r], cs);   // dist minus row-const
            float sc1 = fmaf(-2.0f, acc1[r], cs);
            if (sc0 < best[0][r]) { best[0][r] = sc0; bidx[0][r] = col; }
            if (sc1 < best[1][r]) { best[1][r] = sc1; bidx[1][r] = col; }
        }
    }

    // ---- Cross-lane argmin within each 16-lane half (rows r / r+8),
    // lower index wins ties (matches argmax(-dist) first-hit).
    #pragma unroll
    for (int s = 0; s < 2; ++s) {
        #pragma unroll
        for (int r = 0; r < 8; ++r) {
            #pragma unroll
            for (int m = 8; m >= 1; m >>= 1) {
                float ov = __shfl_xor(best[s][r], m, 32);
                int   oi = __shfl_xor(bidx[s][r], m, 32);
                if (ov < best[s][r] || (ov == best[s][r] && oi < bidx[s][r])) {
                    best[s][r] = ov; bidx[s][r] = oi;
                }
            }
        }
    }

    // ---- Emit indices (as float, per output convention)
    if (l == 0) {
        #pragma unroll
        for (int s = 0; s < 2; ++s)
            #pragma unroll
            for (int r = 0; r < 8; ++r) {
                int row = s * 16 + half * 8 + r;   // lane0 -> rows 0-7, lane16 -> 8-15
                indout[r0 + row] = (float)bidx[s][r];
            }
    }

    // ---- Gather winning code rows (exact f32) + MSE partial
    float ds = 0.0f;
    #pragma unroll
    for (int rr = 0; rr < 32; ++rr) {
        const int e = __shfl(bidx[rr >> 4][rr & 7], ((rr >> 3) & 1) << 4, 32);
        const float* src = embT + (size_t)e * K_DIM + lane * 8;
        const float* xs  = x    + (size_t)(r0 + rr) * K_DIM + lane * 8;
        float*       dst = qout + (size_t)(r0 + rr) * K_DIM + lane * 8;
        v4f q0 = *(const v4f*)(src);
        v4f q1 = *(const v4f*)(src + 4);
        v4f x0 = *(const v4f*)(xs);
        v4f x1 = *(const v4f*)(xs + 4);
        *(v4f*)(dst)     = q0;
        *(v4f*)(dst + 4) = q1;
        #pragma unroll
        for (int i = 0; i < 4; ++i) {
            float d0 = q0[i] - x0[i], d1 = q1[i] - x1[i];
            ds = fmaf(d0, d0, ds);
            ds = fmaf(d1, d1, ds);
        }
    }
    #pragma unroll
    for (int m = 16; m >= 1; m >>= 1) ds += __shfl_xor(ds, m, 32);
    if (lane == 0) partials[blockIdx.x * 8 + wave] = ds;
}

// ---------------------------------------------------------------------------
// Deterministic final reduction of 2048 MSE partials -> diff scalar
// ---------------------------------------------------------------------------
__global__ __launch_bounds__(256) void vq_reduce(
    const float* __restrict__ partials, float* __restrict__ diff)
{
    __shared__ float sm[256];
    int t = threadIdx.x;
    float s = 0.0f;
    for (int i = t; i < 2048; i += 256) s += partials[i];
    sm[t] = s;
    __syncthreads();
    for (int w = 128; w > 0; w >>= 1) {
        if (t < w) sm[t] += sm[t + w];
        __syncthreads();
    }
    if (t == 0) diff[0] = sm[0] * (1.0f / 16777216.0f);
}

extern "C" void kernel_launch(void* const* d_in, const int* in_sizes, int n_in,
                              void* d_out, int out_size, void* d_ws, size_t ws_size,
                              hipStream_t stream)
{
    const float* x     = (const float*)d_in[0];   // [16,4096,256] f32
    const float* embed = (const float*)d_in[1];   // [256,4096] f32
    float* out = (float*)d_out;                   // quantize | diff | indices

    char* ws = (char*)d_ws;
    float*          embT     = (float*)ws;                               // 4 MB
    unsigned short* eb       = (unsigned short*)(ws + (4u << 20));       // 2 MB
    float*          cn       = (float*)(ws + (6u << 20));                // 16 KB
    float*          partials = (float*)(ws + (6u << 20) + (16u << 10));  // 8 KB

    vq_prep_transpose<<<(K_DIM * N_EMB) / 256, 256, 0, stream>>>(embed, embT, eb);
    vq_prep_norm<<<N_EMB / 256, 256, 0, stream>>>(embed, cn);
    vq_main<<<N_TOK / 256, 256, 0, stream>>>(x, eb, cn, embT,
                                             out, out + IND_OFF, partials);
    vq_reduce<<<1, 256, 0, stream>>>(partials, out + DIFF_OFF);
}
